// ModifiedMultiHeadAttention_10737418240805
// MI455X (gfx1250) — matrix operand order repaired and checked by reading, not compiled, not run on hardware
//
#include <hip/hip_runtime.h>
#include <hip/hip_bf16.h>
#include <stdint.h>

// ModifiedMultiHeadAttention (sparsemax attention), MI455X / gfx1250, wave32.
// B=8, L=1024, D=768, H=12, d_head=64.

#define NB   8
#define NL   1024
#define ND   768
#define NH   12
#define DH   64

typedef __attribute__((ext_vector_type(16))) __bf16 v16bf;
typedef __attribute__((ext_vector_type(8)))  float  v8f;

union BF16Frag {
  v16bf v;
  __bf16 e[16];
  uint4 q[2];
};

// Native conversion -> v_cvt_pk_bf16_f32 (round-to-nearest-even).
__device__ __forceinline__ __bf16 f2bf(float f) { return (__bf16)f; }

__device__ __forceinline__ unsigned short f2bfu(float f) {
  union { __bf16 h16; unsigned short u; } cv;
  cv.h16 = (__bf16)f;
  return cv.u;
}

// ---------------------------------------------------------------------------
// Kernel 1: Q/K projections.  out[r,c] = sum_d X[r,d]*W[c,d] + bias[c]
// Both operands are K-contiguous in memory -> direct WMMA A/B fragment loads.
// Results stored bf16, head-major: qh/kh[(b*NH+h)*NL + l][DH]
// One wave per 16x16 output tile; 8 waves per block.
// ---------------------------------------------------------------------------
__global__ void __launch_bounds__(256)
proj_kernel(const float* __restrict__ qg, const float* __restrict__ kg,
            const float* __restrict__ Wq, const float* __restrict__ bq,
            const float* __restrict__ Wk, const float* __restrict__ bk,
            unsigned short* __restrict__ qh, unsigned short* __restrict__ kh) {
  const int lane = threadIdx.x & 31;
  const int wave = threadIdx.x >> 5;
  int tile = blockIdx.x * 8 + wave;              // 2 * 512 * 48 tiles total
  const int which = tile / (512 * 48);           // 0 = Q, 1 = K
  tile -= which * (512 * 48);
  const int rowTile = tile / 48;                 // over B*L/16
  const int colTile = tile - rowTile * 48;       // over D/16

  const float* X    = which ? kg : qg;
  const float* W    = which ? Wk : Wq;
  const float* bias = which ? bk : bq;
  unsigned short* dst = which ? kh : qh;

  const int half = lane >> 4;                    // wave half selects K sub-block
  const int l16  = lane & 15;
  const int arow = rowTile * 16 + l16;           // A fragment row (both halves)
  const int bcol = colTile * 16 + l16;           // B fragment column

  v8f c = {};
  for (int kk = 0; kk < ND; kk += 32) {
    BF16Frag a, bfr;
    // A 16x32: elements 0..7 -> K = kb..kb+7 ; 8..15 -> K = kb+16..kb+23
    const float* ap = X + (size_t)arow * ND + kk + half * 8;
#pragma unroll
    for (int i = 0; i < 8; ++i) a.e[i]     = f2bf(ap[i]);
#pragma unroll
    for (int i = 0; i < 8; ++i) a.e[8 + i] = f2bf(ap[16 + i]);
    // B 32x16: lane = col, same K order as A
    const float* bp = W + (size_t)bcol * ND + kk + half * 8;
#pragma unroll
    for (int i = 0; i < 8; ++i) bfr.e[i] = f2bf(bp[i]);
#pragma unroll
    for (int i = 0; i < 8; ++i) bfr.e[8 + i] = f2bf(bp[16 + i]);
    c = __builtin_amdgcn_wmma_f32_16x16x32_bf16(false, a.v, false, bfr.v,
                                                (short)0, c, false, false);
  }

  const float bv = bias[bcol];
  const int h = bcol >> 6, d = bcol & 63;
#pragma unroll
  for (int j = 0; j < 8; ++j) {                  // C/D: M = j + 8*half, N = l16
    const int r  = rowTile * 16 + j + half * 8;  // flat row in B*L
    const int bb = r >> 10, l = r & 1023;
    dst[(((size_t)(bb * NH + h) << 10) + l) * DH + d] = f2bfu(c[j] + bv);
  }
}

// ---------------------------------------------------------------------------
// Kernel 2: V transpose + bf16 convert:  vt[b][d][l] = bf16(v[b][l][d]).
// Makes the attn@V B-fragments K-contiguous (two 16B loads per fragment
// instead of 16 strided scalar loads).  Coalesced both ways via padded LDS.
// One 32x32 tile per 256-thread block.
// ---------------------------------------------------------------------------
__global__ void __launch_bounds__(256)
vtrans_kernel(const float* __restrict__ vg, unsigned short* __restrict__ vt) {
  __shared__ unsigned short tile[32 * 33];
  int idx = blockIdx.x;                       // NB * 32 * 24 blocks
  const int b  = idx / (32 * 24);
  idx -= b * (32 * 24);
  const int lt = idx / 24;                    // l-tile (32 rows of l)
  const int dt = idx - lt * 24;               // d-tile (32 rows of d)
  const int l0 = lt << 5, d0 = dt << 5;

  const int r = threadIdx.x >> 5;             // 0..7
  const int cidx = threadIdx.x & 31;          // 0..31
#pragma unroll
  for (int rr = 0; rr < 4; ++rr) {
    const int li = r + rr * 8;
    tile[li * 33 + cidx] =
        f2bfu(vg[((size_t)(b * NL) + l0 + li) * ND + d0 + cidx]);
  }
  __syncthreads();
#pragma unroll
  for (int rr = 0; rr < 4; ++rr) {
    const int di = r + rr * 8;
    vt[((size_t)(b * ND) + d0 + di) * NL + l0 + cidx] = tile[cidx * 33 + di];
  }
}

// ---------------------------------------------------------------------------
// Kernel 3: fused scores -> sparsemax -> head-mean -> (attn out) -> attn @ V.
// One 256-thread block (8 waves) per (b, 16-query tile).  LDS: 128 KB
// (16x1024 f32 head-mean accumulator + 16x1024 f32 score tile).
// ---------------------------------------------------------------------------
__global__ void __launch_bounds__(256)
attn_kernel(const unsigned short* __restrict__ qh,
            const unsigned short* __restrict__ kh,
            const unsigned short* __restrict__ vt,  // [B, D, L] bf16
            float* __restrict__ out,     // [B, L, D]
            float* __restrict__ attn) {  // [B, L, L]
  extern __shared__ char smem[];
  float* acc = (float*)smem;             // 16*1024 head-mean accumulator
  float* sc  = acc + 16 * 1024;          // 16*1024 current head's scores

  const int tid  = threadIdx.x;
  const int lane = tid & 31;
  const int wave = tid >> 5;
  const int b    = blockIdx.x >> 6;
  const int q0   = (blockIdx.x & 63) << 4;

  const int half = lane >> 4;
  const int l16  = lane & 15;

  for (int i = tid; i < 16 * 1024; i += 256) acc[i] = 0.0f;
  __syncthreads();

  for (int h = 0; h < NH; ++h) {
    const unsigned short* qhb = qh + ((size_t)(b * NH + h) << 10) * DH;
    const unsigned short* khb = kh + ((size_t)(b * NH + h) << 10) * DH;

    // ---- A fragments (the q-tile) are invariant across all 64 column tiles.
    BF16Frag a0, a1;
    {
      const unsigned short* ap = qhb + (size_t)(q0 + l16) * DH + half * 8;
      a0.q[0] = *(const uint4*)(ap);
      a0.q[1] = *(const uint4*)(ap + 16);
      a1.q[0] = *(const uint4*)(ap + 32);
      a1.q[1] = *(const uint4*)(ap + 48);
    }

    // ---- scores tile [16 x 1024] via WMMA: each wave does 8 column tiles ----
    for (int t = wave; t < 64; t += 8) {
      const int kcol = t << 4;
      v8f c = {};
      BF16Frag bf0, bf1;
      const unsigned short* bp = khb + (size_t)(kcol + l16) * DH + half * 8;
      bf0.q[0] = *(const uint4*)(bp);
      bf0.q[1] = *(const uint4*)(bp + 16);
      bf1.q[0] = *(const uint4*)(bp + 32);
      bf1.q[1] = *(const uint4*)(bp + 48);
      c = __builtin_amdgcn_wmma_f32_16x16x32_bf16(false, a0.v, false, bf0.v,
                                                  (short)0, c, false, false);
      c = __builtin_amdgcn_wmma_f32_16x16x32_bf16(false, a1.v, false, bf1.v,
                                                  (short)0, c, false, false);
#pragma unroll
      for (int j = 0; j < 8; ++j)
        sc[(j + half * 8) * 1024 + kcol + l16] = c[j] * 0.125f;  // 1/sqrt(64)
    }
    __syncthreads();

    // ---- sparsemax per row: 16 lanes per row.  Newton on the convex,
    //      piecewise-linear f(tau) = sum(max(z - tau, 0)) - 1, started from
    //      tau0 = max(z) - 1 (f >= 0 there => monotone convergence to the
    //      exact root; no sort needed).  Row slice cached in registers.
    {
      const int row = tid >> 4;
      const int li  = tid & 15;
      float* srow = sc + row * 1024;

      float z[64];
#pragma unroll
      for (int i = 0; i < 64; ++i) z[i] = srow[li + (i << 4)];

      float m = -3.4e38f;
#pragma unroll
      for (int i = 0; i < 64; ++i) m = fmaxf(m, z[i]);
#pragma unroll
      for (int dd = 1; dd < 16; dd <<= 1) m = fmaxf(m, __shfl_xor(m, dd, 32));

      float tau = m - 1.0f;
      for (int it = 0; it < 24; ++it) {
        float S = 0.0f, cnt = 0.0f;
#pragma unroll
        for (int i = 0; i < 64; ++i) {
          if (z[i] > tau) { S += z[i]; cnt += 1.0f; }
        }
#pragma unroll
        for (int dd = 1; dd < 16; dd <<= 1) {
          S   += __shfl_xor(S,   dd, 32);
          cnt += __shfl_xor(cnt, dd, 32);
        }
        if (cnt > 0.0f) tau = (S - 1.0f) / cnt;
      }

      float* arow = acc + row * 1024;
      const float invH = 1.0f / (float)NH;
#pragma unroll
      for (int i = 0; i < 64; ++i)
        arow[li + (i << 4)] += fmaxf(z[i] - tau, 0.0f) * invH;
    }
    __syncthreads();
  }

  // ---- emit attn (second reference output) ----
  {
    const int row = tid >> 4;
    const int li  = tid & 15;
    const float* arow = acc + row * 1024;
    float* aout = attn + ((size_t)(b * NL) + q0 + row) * (size_t)NL;
    for (int i = 0; i < 64; ++i) aout[li + (i << 4)] = arow[li + (i << 4)];
  }

  // ---- convert head-mean to bf16 (reuse score LDS) for the output GEMM ----
  unsigned short* abf = (unsigned short*)sc;
  for (int i = tid; i < 16 * 1024; i += 256) abf[i] = f2bfu(acc[i]);
  __syncthreads();

  // ---- out[16 x 768] = attn_mean[16 x 1024] @ V[1024 x 768].
  //      B fragment from transposed vt: two runs of 8 contiguous bf16 per lane.
  for (int nt = 0; nt < 6; ++nt) {
    const int ncol = wave * 96 + nt * 16;
    v8f c = {};
    for (int kt = 0; kt < 32; ++kt) {
      BF16Frag a, bf;
      const unsigned short* ap = abf + l16 * 1024 + kt * 32 + half * 8;
      a.q[0] = *(const uint4*)(ap);        // ds_load_b128 from LDS
      a.q[1] = *(const uint4*)(ap + 16);
      const unsigned short* bp =
          vt + ((size_t)(b * ND) + ncol + l16) * NL + kt * 32 + half * 8;
      bf.q[0] = *(const uint4*)(bp);
      bf.q[1] = *(const uint4*)(bp + 16);
      c = __builtin_amdgcn_wmma_f32_16x16x32_bf16(false, a.v, false, bf.v,
                                                  (short)0, c, false, false);
    }
#pragma unroll
    for (int j = 0; j < 8; ++j)
      out[((size_t)(b * NL) + q0 + j + half * 8) * ND + ncol + l16] = c[j];
  }
}

// ---------------------------------------------------------------------------
extern "C" void kernel_launch(void* const* d_in, const int* in_sizes, int n_in,
                              void* d_out, int out_size, void* d_ws, size_t ws_size,
                              hipStream_t stream) {
  (void)in_sizes; (void)n_in; (void)out_size; (void)ws_size;
  const float* q  = (const float*)d_in[0];
  const float* k  = (const float*)d_in[1];
  const float* v  = (const float*)d_in[2];
  const float* Wq = (const float*)d_in[3];
  const float* bq = (const float*)d_in[4];
  const float* Wk = (const float*)d_in[5];
  const float* bk = (const float*)d_in[6];

  float* out  = (float*)d_out;                       // [8,1024,768]
  float* attn = out + (size_t)NB * NL * ND;          // [8,1024,1024]

  unsigned short* qh = (unsigned short*)d_ws;                    // 12.6 MB bf16
  unsigned short* kh = qh + (size_t)NB * NH * NL * DH;           // 12.6 MB bf16
  unsigned short* vt = kh + (size_t)NB * NH * NL * DH;           // 12.6 MB bf16

  // Stage 1: Q/K projections (2 * 512 * 48 = 49152 wave-tiles, 8 waves/block)
  proj_kernel<<<6144, 256, 0, stream>>>(q, k, Wq, bq, Wk, bk, qh, kh);

  // Stage 1b: V -> bf16 transpose (independent of stage 1; overlaps with it)
  vtrans_kernel<<<NB * 32 * 24, 256, 0, stream>>>(v, vt);

  // Stage 2: fused scores/sparsemax/mean/output.  128 KB dynamic LDS per WG
  // (CDNA5 WGP has 320 KB, so 2 workgroups can co-reside).
  const size_t smem = (size_t)2 * 16 * 1024 * sizeof(float);
  hipFuncSetAttribute((const void*)attn_kernel,
                      hipFuncAttributeMaxDynamicSharedMemorySize, (int)smem);
  attn_kernel<<<NB * (NL / 16), 256, smem, stream>>>(qh, kh, vt, out, attn);
}
